// GRUPredictor_74543452389720
// MI455X (gfx1250) — compile-verified, matrix operand order checked by reading, not run on hardware
//
#include <hip/hip_runtime.h>

typedef __attribute__((ext_vector_type(16))) _Float16 v16h;
typedef __attribute__((ext_vector_type(8)))  _Float16 v8h;
typedef __attribute__((ext_vector_type(8)))  float    v8f;

#define NW        4                     // waves per block
#define TLEN      512
#define TPAD      520                   // padded floats per x row (keeps 16B align, spreads banks)
#define FRAG_B    1024                  // one B fragment: 32 lanes * 32 bytes
#define NFRAG     72                    // 3 matrices * 12 n-tiles * 2 k-halves
#define LDS_X_OFF   (NFRAG * FRAG_B)            // 73728
#define LDS_X_WAVE  (16 * TPAD * 4)             // 33280
#define LDS_H_OFF   (LDS_X_OFF + NW * LDS_X_WAVE)
#define LDS_H_WAVE  4096                        // hA1(2KB) + hA2(2KB), f16 [16 rows][64 k]
#define LDS_TOTAL   (LDS_H_OFF + NW * LDS_H_WAVE)

// ---- fast nonlinearities: native v_tanh_f32 if available (TRANS op, co-executes
// ---- with XDL WMMAs), else exp + raw v_rcp (no precise-division refinement chain)
__device__ __forceinline__ float tanh_fast(float v) {
#if __has_builtin(__builtin_amdgcn_tanhf)
  return __builtin_amdgcn_tanhf(v);
#else
  float e = __expf(-2.0f * v);
  return (1.0f - e) * __builtin_amdgcn_rcpf(1.0f + e);
#endif
}
__device__ __forceinline__ float sigm_fast(float v) {
  return __builtin_fmaf(0.5f, tanh_fast(0.5f * v), 0.5f);
}

__device__ __forceinline__ v8f wmma_f16(v16h a, v16h b, v8f c) {
  // v_wmma_f32_16x16x32_f16: D = A(16x32 f16) * B(32x16 f16) + C(16x16 f32)
  return __builtin_amdgcn_wmma_f32_16x16x32_f16(false, a, false, b, (short)0, c, false, false);
}

// B fragment: stored contiguously per lane at staging time (32B/lane)
__device__ __forceinline__ v16h load_bfrag(const char* smem, int mat, int nt, int kh, int lane) {
  return *(const v16h*)(smem + (mat * 24 + nt * 2 + kh) * FRAG_B + lane * 32);
}

// A fragment from hA buffer (f16 [16 rows][64 k], 128B row stride), per ISA 16-bit A layout:
// lane l (col=l&15, hi=l>>4): halves 0..7 -> k = kh*32 + hi*8 + j ; halves 8..15 -> +16
__device__ __forceinline__ v16h load_afrag(const char* hbuf, int kh, int col, int hi) {
  const char* p = hbuf + col * 128 + kh * 64 + hi * 16;
  v8h lo = *(const v8h*)p;
  v8h hh = *(const v8h*)(p + 32);
  return __builtin_shufflevector(lo, hh, 0,1,2,3,4,5,6,7,8,9,10,11,12,13,14,15);
}

__global__ __launch_bounds__(NW * 32)
void gru2_fused_wmma(const float* __restrict__ x,
                     const float* __restrict__ Wih0, const float* __restrict__ Whh0,
                     const float* __restrict__ bih0, const float* __restrict__ bhh0,
                     const float* __restrict__ Wih1, const float* __restrict__ Whh1,
                     const float* __restrict__ bih1, const float* __restrict__ bhh1,
                     const float* __restrict__ fcw,  const float* __restrict__ fcb,
                     float* __restrict__ out) {
  extern __shared__ char smem[];
  const int tid = threadIdx.x;
  const int w   = tid >> 5;        // wave in block
  const int l   = tid & 31;        // lane
  const int col = l & 15;
  const int hi  = l >> 4;
  const int b0  = blockIdx.x * (NW * 16) + w * 16;   // batch base for this wave

  // ---- one-time: pack weight matrices into LDS as f16 B-fragments ----
  // B[k][n] = W[n][k]; lane l holds column n = nt*16+col.
  const float* mats[3] = { Whh0, Wih1, Whh1 };
  for (int f = w; f < NFRAG; f += NW) {
    const int mat = f / 24, rem = f % 24, nt = rem >> 1, kh = rem & 1;
    const float* W = mats[mat];
    const int n = nt * 16 + col;
    v16h vals;
#pragma unroll
    for (int j = 0; j < 8; ++j) vals[j]     = (_Float16)W[n * 64 + kh * 32 + hi * 8 + j];
#pragma unroll
    for (int j = 0; j < 8; ++j) vals[8 + j] = (_Float16)W[n * 64 + kh * 32 + 16 + hi * 8 + j];
    *(v16h*)(smem + f * FRAG_B + l * 32) = vals;
  }

  // ---- one-time: stage this wave's x tile (16 rows x 512 t, f32) ----
  char* xw = smem + LDS_X_OFF + w * LDS_X_WAVE;
  for (int r = 0; r < 16; ++r) {
    const float4* src = (const float4*)(x + (b0 + r) * TLEN);
    float4* dst = (float4*)(xw + r * (TPAD * 4));
#pragma unroll
    for (int i = 0; i < 4; ++i) dst[i * 32 + l] = src[i * 32 + l];
  }

  // ---- zero h A-layout buffers (h0 = 0) ----
  char* hA1 = smem + LDS_H_OFF + w * LDS_H_WAVE;
  char* hA2 = hA1 + 2048;
  {
    float4 z4 = {0.f, 0.f, 0.f, 0.f};
#pragma unroll
    for (int i = 0; i < 8; ++i) ((float4*)hA1)[i * 32 + l] = z4;   // covers hA1+hA2 (4KB)
  }
  __syncthreads();

  // ---- per-lane bias / tiny-weight registers (gate order: r=0..3, z=4..7, n=8..11 tiles) ----
  float w0rz[8], b0rz[8], b1rz[8];
  float w0n[4], b0in[4], b0hn[4], b1in[4], b1hn[4];
#pragma unroll
  for (int q = 0; q < 8; ++q) {
    const int g = q * 16 + col;
    w0rz[q] = Wih0[g];
    b0rz[q] = bih0[g] + bhh0[g];
    b1rz[q] = bih1[g] + bhh1[g];
  }
#pragma unroll
  for (int q = 0; q < 4; ++q) {
    const int g = (q + 8) * 16 + col;
    w0n[q]  = Wih0[g];
    b0in[q] = bih0[g];
    b0hn[q] = bhh0[g];
    b1in[q] = bih1[g];
    b1hn[q] = bhh1[g];
  }

  // recurrent state in C-layout registers: h[tile gi][vgpr v] <-> (row v+8*hi, col gi*16+col)
  float h1c[4][8], h2c[4][8];
#pragma unroll
  for (int gi = 0; gi < 4; ++gi)
#pragma unroll
    for (int v = 0; v < 8; ++v) { h1c[gi][v] = 0.f; h2c[gi][v] = 0.f; }

#pragma unroll 1
  for (int t = 0; t < TLEN; ++t) {
    // stop LLVM from hoisting loop-invariant LDS B-fragments into (far too many) VGPRs
    __asm__ __volatile__("" ::: "memory");

    float xr[8];
#pragma unroll
    for (int v = 0; v < 8; ++v)
      xr[v] = *(const float*)(xw + (v + 8 * hi) * (TPAD * 4) + t * 4);

    const v16h a1k0 = load_afrag(hA1, 0, col, hi), a1k1 = load_afrag(hA1, 1, col, hi);
    const v16h a2k0 = load_afrag(hA2, 0, col, hi), a2k1 = load_afrag(hA2, 1, col, hi);

    // ================= layer 0 =================
#pragma unroll
    for (int gi = 0; gi < 4; ++gi) {
      v8f racc, zacc, hnacc;
#pragma unroll
      for (int v = 0; v < 8; ++v) {
        racc[v]  = __builtin_fmaf(xr[v], w0rz[gi],     b0rz[gi]);      // x-gate + b_ih + b_hh (r)
        zacc[v]  = __builtin_fmaf(xr[v], w0rz[gi + 4], b0rz[gi + 4]);  // (z)
        hnacc[v] = b0hn[gi];                                           // h-gate bias only (n)
      }
      racc  = wmma_f16(a1k0, load_bfrag(smem, 0, gi,     0, l), racc);
      racc  = wmma_f16(a1k1, load_bfrag(smem, 0, gi,     1, l), racc);
      zacc  = wmma_f16(a1k0, load_bfrag(smem, 0, gi + 4, 0, l), zacc);
      zacc  = wmma_f16(a1k1, load_bfrag(smem, 0, gi + 4, 1, l), zacc);
      hnacc = wmma_f16(a1k0, load_bfrag(smem, 0, gi + 8, 0, l), hnacc);
      hnacc = wmma_f16(a1k1, load_bfrag(smem, 0, gi + 8, 1, l), hnacc);
#pragma unroll
      for (int v = 0; v < 8; ++v) {
        const float r  = sigm_fast(racc[v]);
        const float z  = sigm_fast(zacc[v]);
        const float xn = __builtin_fmaf(xr[v], w0n[gi], b0in[gi]);
        const float n  = tanh_fast(__builtin_fmaf(r, hnacc[v], xn));
        h1c[gi][v] = __builtin_fmaf(z, h1c[gi][v] - n, n);   // (1-z)*n + z*h
      }
    }
    // store h1_new to hA1 (f16, A-layout-friendly [row][k])
#pragma unroll
    for (int gi = 0; gi < 4; ++gi)
#pragma unroll
      for (int v = 0; v < 8; ++v)
        ((_Float16*)hA1)[(v + 8 * hi) * 64 + gi * 16 + col] = (_Float16)h1c[gi][v];

    const v16h a1nk0 = load_afrag(hA1, 0, col, hi), a1nk1 = load_afrag(hA1, 1, col, hi);

    // ================= layer 1 =================
#pragma unroll
    for (int gi = 0; gi < 4; ++gi) {
      v8f racc, zacc, inacc, hnacc;
#pragma unroll
      for (int v = 0; v < 8; ++v) {
        racc[v]  = b1rz[gi];
        zacc[v]  = b1rz[gi + 4];
        inacc[v] = b1in[gi];
        hnacc[v] = b1hn[gi];
      }
      // r,z: fused K=128 over [h1_new | h2_old]
      racc  = wmma_f16(a1nk0, load_bfrag(smem, 1, gi,     0, l), racc);
      racc  = wmma_f16(a1nk1, load_bfrag(smem, 1, gi,     1, l), racc);
      racc  = wmma_f16(a2k0,  load_bfrag(smem, 2, gi,     0, l), racc);
      racc  = wmma_f16(a2k1,  load_bfrag(smem, 2, gi,     1, l), racc);
      zacc  = wmma_f16(a1nk0, load_bfrag(smem, 1, gi + 4, 0, l), zacc);
      zacc  = wmma_f16(a1nk1, load_bfrag(smem, 1, gi + 4, 1, l), zacc);
      zacc  = wmma_f16(a2k0,  load_bfrag(smem, 2, gi + 4, 0, l), zacc);
      zacc  = wmma_f16(a2k1,  load_bfrag(smem, 2, gi + 4, 1, l), zacc);
      // n: x-part and h-part kept separate (r gates only the h-part)
      inacc = wmma_f16(a1nk0, load_bfrag(smem, 1, gi + 8, 0, l), inacc);
      inacc = wmma_f16(a1nk1, load_bfrag(smem, 1, gi + 8, 1, l), inacc);
      hnacc = wmma_f16(a2k0,  load_bfrag(smem, 2, gi + 8, 0, l), hnacc);
      hnacc = wmma_f16(a2k1,  load_bfrag(smem, 2, gi + 8, 1, l), hnacc);
#pragma unroll
      for (int v = 0; v < 8; ++v) {
        const float r = sigm_fast(racc[v]);
        const float z = sigm_fast(zacc[v]);
        const float n = tanh_fast(__builtin_fmaf(r, hnacc[v], inacc[v]));
        h2c[gi][v] = __builtin_fmaf(z, h2c[gi][v] - n, n);   // (1-z)*n + z*h
      }
    }
    // store h2_new to hA2 for next timestep's A-fragments
#pragma unroll
    for (int gi = 0; gi < 4; ++gi)
#pragma unroll
      for (int v = 0; v < 8; ++v)
        ((_Float16*)hA2)[(v + 8 * hi) * 64 + gi * 16 + col] = (_Float16)h2c[gi][v];
  }

  // ---- FC head: out[b] = dot(h2[b,:], fc_w) + fc_b ----
  // (loaded only here so fw/fb don't occupy registers across the 512-step loop)
  float fw[4];
#pragma unroll
  for (int q = 0; q < 4; ++q) fw[q] = fcw[q * 16 + col];
  const float fb = fcb[0];
#pragma unroll
  for (int v = 0; v < 8; ++v) {
    float s = 0.f;
#pragma unroll
    for (int gi = 0; gi < 4; ++gi) s += h2c[gi][v] * fw[gi];
    s += __shfl_xor(s, 1, 32);
    s += __shfl_xor(s, 2, 32);
    s += __shfl_xor(s, 4, 32);
    s += __shfl_xor(s, 8, 32);
    if (col == 0) out[b0 + v + 8 * hi] = s + fb;
  }
}

extern "C" void kernel_launch(void* const* d_in, const int* in_sizes, int n_in,
                              void* d_out, int out_size, void* d_ws, size_t ws_size,
                              hipStream_t stream) {
  (void)in_sizes; (void)n_in; (void)out_size; (void)d_ws; (void)ws_size;
  const float* x    = (const float*)d_in[0];
  const float* Wih0 = (const float*)d_in[1];
  const float* Whh0 = (const float*)d_in[2];
  const float* bih0 = (const float*)d_in[3];
  const float* bhh0 = (const float*)d_in[4];
  const float* Wih1 = (const float*)d_in[5];
  const float* Whh1 = (const float*)d_in[6];
  const float* bih1 = (const float*)d_in[7];
  const float* bhh1 = (const float*)d_in[8];
  const float* fcw  = (const float*)d_in[9];
  const float* fcb  = (const float*)d_in[10];

  hipFuncSetAttribute((const void*)gru2_fused_wmma,
                      hipFuncAttributeMaxDynamicSharedMemorySize, LDS_TOTAL);

  const int B = 2048;
  dim3 grid(B / (NW * 16));   // 32 blocks * 4 waves * 16 batch rows
  dim3 block(NW * 32);
  gru2_fused_wmma<<<grid, block, LDS_TOTAL, stream>>>(
      x, Wih0, Whh0, bih0, bhh0, Wih1, Whh1, bih1, bhh1, fcw, fcb, (float*)d_out);
}
